// CustomFullyConnectedLayer_11690900980058
// MI455X (gfx1250) — matrix operand order repaired and checked by reading, not compiled, not run on hardware
//
#include <hip/hip_runtime.h>

// ---------------- problem constants ----------------
#define N_F     8192        // IN_F == OUT_F
#define NBATCH  1024
#define TOPK_K  820.0f      // ceil(0.1 * 8192)
#define TOPK_L  0.05f
#define DYK_IT  50

// ---------------- GEMM tiling ----------------
#define BM      128
#define BN      128
#define BK      32
#define BAND    160         // 159 diagonals needed per (128 x 32) W tile, round up
#define PITCH   36          // LDS pitch (bf16): 72B rows -> b64 stores, 18-dword lane stride (conflict-free per 16-lane group)
#define NKT     (N_F / BK)  // 256 K-steps
// byte-offset stepping for V diagonal walk: d*32768 mod 2^28 == (d mod 8192)*32768
#define VMASKB  0xFFFFFFFu  // 2^28 - 1
#define VSTEPB  0xFF00000u  // 2^28 - 32*32768 : advance d by -32 rows (mod 8192), in bytes

typedef __attribute__((ext_vector_type(16))) __bf16 v16bf;
typedef __attribute__((ext_vector_type(8)))  float  v8f;

// pack two f32 -> packed bf16 pair (round-to-nearest, ties-away): 2 adds + 1 v_perm
static __device__ __forceinline__ unsigned pack_bf16(float f0, float f1) {
    union { float f; unsigned u; } a, b; a.f = f0; b.f = f1;
    const unsigned u0 = a.u + 0x8000u;
    const unsigned u1 = b.u + 0x8000u;
    return __builtin_amdgcn_perm(u1, u0, 0x07060302);   // {u1[31:16], u0[31:16]}
}

// =====================================================================
// Kernel 1: Dykstra projection -> soft top-k mask a (8192 f32 into d_ws)
// =====================================================================
__global__ void __launch_bounds__(1024)
dykstra_topk_kernel(const float* __restrict__ alpha, float* __restrict__ aout) {
    __shared__ float red[33];
    const int tid  = threadIdx.x;
    const int lane = tid & 31;
    const int wid  = tid >> 5;

    float v[8], p[8], q[8];
    #pragma unroll
    for (int j = 0; j < 8; ++j) {
        v[j] = alpha[tid + j * 1024] * (1.0f / TOPK_L);
        p[j] = 0.0f; q[j] = 0.0f;
    }

    for (int it = 0; it < DYK_IT; ++it) {
        float t[8];
        float local = 0.0f;
        #pragma unroll
        for (int j = 0; j < 8; ++j) { t[j] = v[j] + p[j]; local += t[j]; }
        #pragma unroll
        for (int off = 16; off > 0; off >>= 1) local += __shfl_xor(local, off, 32);
        if (lane == 0) red[wid] = local;
        __syncthreads();
        if (tid == 0) {
            float s = 0.0f;
            for (int i = 0; i < 32; ++i) s += red[i];
            red[32] = s;
        }
        __syncthreads();
        const float adj = (TOPK_K - red[32]) * (1.0f / (float)N_F);
        #pragma unroll
        for (int j = 0; j < 8; ++j) {
            const float y  = t[j] + adj;                    // hyperplane projection
            p[j] = t[j] - y;
            const float z  = y + q[j];
            const float vn = fminf(fmaxf(z, 0.0f), 1.0f);   // box projection
            v[j] = vn;
            q[j] = z - vn;
        }
    }
    #pragma unroll
    for (int j = 0; j < 8; ++j) aout[tid + j * 1024] = v[j];
}

// =====================================================================
// Kernel 2: out[b,r] = sum_c x[b,c] * a[(r-c)%N] * V[(r-c)%N, c]
// Diagonal-band LDS staging + v_wmma_f32_16x16x32_bf16, double-buffered,
// incremental 32-bit *byte* offsets so loads use the GVS saddr+voffset form.
// Block: 256 threads (8 waves, 2 M-groups x 4 N-groups), tile 128x128x(K=32).
// =====================================================================
__global__ void __launch_bounds__(256)
diag_gemm_kernel(const float* __restrict__ x, const float* __restrict__ V,
                 const float* __restrict__ atopk, float* __restrict__ out) {
    __shared__ __align__(16) unsigned short wlds[2][BAND * PITCH];
    __shared__ __align__(16) unsigned short xlds[2][BM * PITCH];

    const int tid    = threadIdx.x;
    const int lane   = tid & 31;
    const int wv     = tid >> 5;
    const int m0     = blockIdx.x * BM;   // batch block (x fastest -> same-r0 blocks share band in L2)
    const int r0     = blockIdx.y * BN;   // out-feature block
    const int wmBase = (wv & 1) * 64;
    const int wnBase = (wv >> 1) * 32;
    const int lm     = lane & 15;
    const int hl     = lane >> 4;

    // fixed per-thread staging coordinates: quad column + base row, rows stride 32
    const int qd = (tid & 7) * 4;
    const int ib = tid >> 3;

    v8f acc[4][2];
    #pragma unroll
    for (int mt = 0; mt < 4; ++mt)
        #pragma unroll
        for (int nt = 0; nt < 2; ++nt)
            #pragma unroll
            for (int e = 0; e < 8; ++e) acc[mt][nt][e] = 0.0f;

    // ---- incremental 32-bit BYTE offsets (loads lower to saddr + u32 voffset) ----
    unsigned voffB[5];                     // d*32768, d = (r0 - 31 + ib + 32u - 32*kt) mod 8192
    #pragma unroll
    for (int u = 0; u < 5; ++u)
        voffB[u] = ((unsigned)(r0 - 31 + ib + 32 * u) & (N_F - 1)) << 15;
    unsigned xoffB[4];                     // (m0+ib+32u)*32768
    #pragma unroll
    for (int u = 0; u < 4; ++u)
        xoffB[u] = (unsigned)(m0 + ib + 32 * u) << 15;
    unsigned colB = (unsigned)qd * 4u;     // (c0 + qd)*4 bytes

    // prefetch registers
    float4 pw[5]; float pa[5]; float4 px[4];

    auto loadG = [&]() {
        #pragma unroll
        for (int u = 0; u < 5; ++u) {
            pa[u] = atopk[voffB[u] >> 15];  // zext index -> global_load_b32 ... scale_offset
            pw[u] = *reinterpret_cast<const float4*>(
                        reinterpret_cast<const char*>(V) + (size_t)(voffB[u] + colB));
        }
        #pragma unroll
        for (int u = 0; u < 4; ++u)
            px[u] = *reinterpret_cast<const float4*>(
                        reinterpret_cast<const char*>(x) + (size_t)(xoffB[u] + colB));
    };
    auto advance = [&]() {
        #pragma unroll
        for (int u = 0; u < 5; ++u) voffB[u] = (voffB[u] + VSTEPB) & VMASKB;
        colB += BK * 4;
    };

    auto storeL = [&](int buf) {
        #pragma unroll
        for (int u = 0; u < 5; ++u) {
            const float a = pa[u];
            uint2 pk;
            pk.x = pack_bf16(pw[u].x * a, pw[u].y * a);
            pk.y = pack_bf16(pw[u].z * a, pw[u].w * a);
            *reinterpret_cast<uint2*>(&wlds[buf][(ib + 32 * u) * PITCH + qd]) = pk;
        }
        #pragma unroll
        for (int u = 0; u < 4; ++u) {
            uint2 pk;
            pk.x = pack_bf16(px[u].x, px[u].y);
            pk.y = pack_bf16(px[u].z, px[u].w);
            *reinterpret_cast<uint2*>(&xlds[buf][(ib + 32 * u) * PITCH + qd]) = pk;
        }
    };

    union FragU { v16bf v; unsigned short u[16]; unsigned long long q[4]; };

    auto compute = [&](int buf) {
        // A fragments (x side), 16x32 bf16: lane-half interleaves K in chunks of 8
        FragU xf[4];
        #pragma unroll
        for (int mt = 0; mt < 4; ++mt) {
            const unsigned short* bp = &xlds[buf][(wmBase + mt * 16 + lm) * PITCH + 8 * hl];
            xf[mt].q[0] = *reinterpret_cast<const unsigned long long*>(bp);
            xf[mt].q[1] = *reinterpret_cast<const unsigned long long*>(bp + 4);
            xf[mt].q[2] = *reinterpret_cast<const unsigned long long*>(bp + 16);
            xf[mt].q[3] = *reinterpret_cast<const unsigned long long*>(bp + 20);
        }
        // B fragments (W side), 32x16 bf16: lane = column n, lane-half interleaves K in chunks of 8.
        // Diagonal skew resolved here: W[r0+rl, c0+k] = band row (rl + 31 - k), col k.
        FragU wf[2];
        #pragma unroll
        for (int nt = 0; nt < 2; ++nt) {
            const int rl = wnBase + nt * 16 + lm;
            #pragma unroll
            for (int j = 0; j < 16; ++j) {
                const int k = (j < 8) ? (j + 8 * hl) : (8 + j + 8 * hl);
                wf[nt].u[j] = wlds[buf][(rl + 31 - k) * PITCH + k];
            }
        }
        #pragma unroll
        for (int mt = 0; mt < 4; ++mt)
            #pragma unroll
            for (int nt = 0; nt < 2; ++nt)
                acc[mt][nt] = __builtin_amdgcn_wmma_f32_16x16x32_bf16(
                    false, xf[mt].v, false, wf[nt].v,
                    (short)0, acc[mt][nt], false, false);
    };

    // ---- software pipeline: prefetch kt+1 while computing kt; 1 barrier/step ----
    loadG(); advance();            // fetch kt = 0
    storeL(0);
    __syncthreads();
    for (int kt = 0; kt < NKT - 2; kt += 2) {   // guard-free steady state
        loadG(); advance();        // fetch kt+1
        compute(0);
        storeL(1);
        __syncthreads();

        loadG(); advance();        // fetch kt+2
        compute(1);
        storeL(0);
        __syncthreads();
    }
    // epilogue: buffer0 holds kt = NKT-2; fetch NKT-1, no further prefetch
    loadG();
    compute(0);
    storeL(1);
    __syncthreads();
    compute(1);

    // ---- store D (f32 16x16: lane n = lane&15, VGPR i -> M = i + 8*(lane>>4)) ----
    #pragma unroll
    for (int mt = 0; mt < 4; ++mt)
        #pragma unroll
        for (int nt = 0; nt < 2; ++nt) {
            const int colo = r0 + wnBase + nt * 16 + lm;
            #pragma unroll
            for (int i = 0; i < 8; ++i) {
                const int row = m0 + wmBase + mt * 16 + i + 8 * hl;
                out[(size_t)row * N_F + colo] = acc[mt][nt][i];
            }
        }
}

// =====================================================================
extern "C" void kernel_launch(void* const* d_in, const int* in_sizes, int n_in,
                              void* d_out, int out_size, void* d_ws, size_t ws_size,
                              hipStream_t stream) {
    (void)in_sizes; (void)n_in; (void)out_size; (void)ws_size;
    const float* x     = (const float*)d_in[0];   // [1024, 8192]
    const float* V     = (const float*)d_in[1];   // [8192, 8192]
    const float* alpha = (const float*)d_in[2];   // [8192]
    float* out   = (float*)d_out;                 // [1024, 8192]
    float* atopk = (float*)d_ws;                  // 8192 f32 scratch (32 KB)

    dykstra_topk_kernel<<<1, 1024, 0, stream>>>(alpha, atopk);
    dim3 grid(NBATCH / BM, N_F / BN);             // (8, 64)
    diag_gemm_kernel<<<grid, 256, 0, stream>>>(x, V, atopk, out);
}
